// BiLSTMModel_22703197127119
// MI455X (gfx1250) — compile-verified, matrix operand order checked by reading, not run on hardware
//
#include <hip/hip_runtime.h>

typedef __attribute__((ext_vector_type(16))) _Float16 v16h;
typedef __attribute__((ext_vector_type(8)))  _Float16 v8h;
typedef __attribute__((ext_vector_type(8)))  float    v8f;

#define S_LEN 512
#define BATCH 256
#define EMB   100
#define HID   64
#define NG    256            // 4*H
#define KPAD  128            // EMB padded to WMMA K multiple
#define ROWS_TOTAL (S_LEN * BATCH)   // 131072

// -------- workspace layout (bytes) --------
#define OFF_GPRE 0ull
#define SZ_GPRE  (2ull * ROWS_TOTAL * NG * 2)          // 134217728 (f16)
#define OFF_WIH  (OFF_GPRE + SZ_GPRE)
#define SZ_WIH   (2ull * NG * KPAD * 2)                // 131072
#define OFF_WHH  (OFF_WIH + SZ_WIH)
#define SZ_WHH   (2ull * NG * HID * 2)                 // 65536
#define OFF_BIAS (OFF_WHH + SZ_WHH)
#define SZ_BIAS  (2ull * NG * 4)                       // 2048
#define OFF_HFIN (OFF_BIAS + SZ_BIAS)                  // 2*256*64 f32

// Load one 16-lane x K32 f16 WMMA fragment. Caller passes pointer to
// element [row][kk + kbase] where kbase = (lane>=16 ? 8 : 0).
// ISA layout: lane<16 -> halves {K0..7, K16..23}; lane>=16 -> {K8..15, K24..31}.
__device__ __forceinline__ v16h load_frag(const _Float16* p) {
    v8h lo = *(const v8h*)(p);
    v8h hi = *(const v8h*)(p + 16);
    v16h r;
#pragma unroll
    for (int i = 0; i < 8; ++i) { r[i] = lo[i]; r[i + 8] = hi[i]; }
    return r;
}

__device__ __forceinline__ v8f wmma_f16(v16h a, v16h b, v8f c) {
    return __builtin_amdgcn_wmma_f32_16x16x32_f16(false, a, false, b,
                                                  (short)0, c, false, false);
}

// Async global -> LDS copy of 16 bytes (CDNA5 GLOBAL_LOAD_ASYNC_TO_LDS_B128,
// tracked by ASYNCcnt). lds_off = byte address within LDS (generic-ptr low 32).
__device__ __forceinline__ void async_ld_b128(unsigned lds_off, const void* gaddr) {
    asm volatile("global_load_async_to_lds_b128 %0, %1, off"
                 :: "v"(lds_off), "v"(gaddr) : "memory");
}
__device__ __forceinline__ void wait_async0() {
    asm volatile("s_wait_asynccnt 0x0" ::: "memory");
}

__device__ __forceinline__ float sigf(float x) {
    return 1.f / (1.f + __expf(-x));
}
__device__ __forceinline__ float tanh_fast(float x) {
    float e = __expf(-2.f * fabsf(x));
    float r = (1.f - e) / (1.f + e);
    return copysignf(r, x);
}

// ---------------- prep: f16 weight staging + fused bias ----------------
__global__ void prep_kernel(const float* Wih_f, const float* Whh_f,
                            const float* bih_f, const float* bhh_f,
                            const float* Wih_b, const float* Whh_b,
                            const float* bih_b, const float* bhh_b,
                            char* ws) {
    _Float16* wih16 = (_Float16*)(ws + OFF_WIH);
    _Float16* whh16 = (_Float16*)(ws + OFF_WHH);
    float*    biasg = (float*)(ws + OFF_BIAS);
    for (int i = threadIdx.x; i < 2 * NG; i += blockDim.x) {
        int dir = i >> 8, n = i & 255;
        const float* Wi = dir ? Wih_b : Wih_f;
        const float* Wh = dir ? Whh_b : Whh_f;
        for (int k = 0; k < EMB; ++k)
            wih16[(size_t)i * KPAD + k] = (_Float16)Wi[n * EMB + k];
        for (int k = EMB; k < KPAD; ++k)
            wih16[(size_t)i * KPAD + k] = (_Float16)0.f;
        for (int k = 0; k < HID; ++k)
            whh16[(size_t)i * HID + k] = (_Float16)Wh[n * HID + k];
        biasg[i] = dir ? (bih_b[n] + bhh_b[n]) : (bih_f[n] + bhh_f[n]);
    }
}

// ---------------- proj: Gpre = gather(emb, x) @ W_ih^T + bias ----------------
// grid = 2048 blocks (dir = blk>>10, 1024 row-blocks of 128 rows), 256 threads.
__global__ void proj_kernel(const int* __restrict__ x,
                            const float* __restrict__ emb,
                            char* ws) {
    __shared__ _Float16 As[128 * KPAD];   // 32 KB gathered f16 embeddings
    const _Float16* wih16 = (const _Float16*)(ws + OFF_WIH);
    const float*    biasg = (const float*)(ws + OFF_BIAS);
    _Float16*       gpre  = (_Float16*)(ws + OFF_GPRE);

    int dir = blockIdx.x >> 10;
    int rb  = blockIdx.x & 1023;
    int R0  = rb * 128;
    int tid = threadIdx.x;

    {   // gather + f32->f16 convert: 2 threads per row
        int r = tid >> 1, p = tid & 1;
        int row = R0 + r;
        int s = row >> 8, b = row & 255;
        int token = x[b * S_LEN + s];
        const float* er = emb + (size_t)token * EMB + p * 50;
        _Float16* dst = As + r * KPAD + p * 50;
        for (int k = 0; k < 50; ++k) dst[k] = (_Float16)er[k];
        if (p) {
            for (int k = EMB; k < KPAD; ++k) As[r * KPAD + k] = (_Float16)0.f;
        }
    }
    __syncthreads();

    int wave = tid >> 5, lane = tid & 31;
    int ln = lane & 15, kb = (lane >> 4) * 8;   // kb==8 <=> upper half-wave
    const _Float16* Arow = As + (wave * 16 + ln) * KPAD + kb;

    v16h afrag[4];
#pragma unroll
    for (int c = 0; c < 4; ++c) afrag[c] = load_frag(Arow + c * 32);

    for (int nt = 0; nt < 16; ++nt) {
        int ncol = nt * 16 + ln;
        float bv = biasg[dir * NG + ncol];
        v8f acc;
#pragma unroll
        for (int r = 0; r < 8; ++r) acc[r] = bv;
        const _Float16* Brow = wih16 + ((size_t)dir * NG + ncol) * KPAD + kb;
#pragma unroll
        for (int c = 0; c < 4; ++c) {
            v16h bf = load_frag(Brow + c * 32);
            acc = wmma_f16(afrag[c], bf, acc);
        }
        size_t base = (size_t)dir * ROWS_TOTAL * NG +
                      (size_t)(R0 + wave * 16) * NG + ncol;
#pragma unroll
        for (int r = 0; r < 8; ++r)
            gpre[base + (size_t)(r + kb) * NG] = (_Float16)acc[r];
    }
}

// ---------------- rec: the LSTM recurrence, batch-split ----------------
// grid = 16 blocks: dir = blk>>3, batch block of 32 rows. 256 threads = 8 waves.
// W_hh^T fragments live in VGPRs; Gpre step tiles are double-buffered in LDS
// via async global->LDS copies overlapped with the current step's WMMA.
__global__ void rec_kernel(char* ws) {
    __shared__ _Float16 hb_s[32 * HID];        // 4 KB  current h (f16)
    __shared__ _Float16 gbuf[2][32 * NG];      // 2 x 16 KB staged Gpre tiles

    const _Float16* whh16 = (const _Float16*)(ws + OFF_WHH);
    const _Float16* gpre  = (const _Float16*)(ws + OFF_GPRE);
    float*          hfin  = (float*)(ws + OFF_HFIN);

    int dir = blockIdx.x >> 3;
    int bb  = blockIdx.x & 7;
    int b0  = bb * 32;
    int tid = threadIdx.x;

    int wave = tid >> 5, lane = tid & 31;
    int m = wave & 1, hbk = wave >> 1;          // m: batch tile, hbk: h col block
    int ln = lane & 15, kb = (lane >> 4) * 8;

    // W_hh^T B-fragments: persistent in VGPRs for all 512 steps
    v16h bfr[8];
#pragma unroll
    for (int gi = 0; gi < 4; ++gi) {
        const _Float16* Brow = whh16 + (size_t)dir * NG * HID +
                               (size_t)(gi * 64 + hbk * 16 + ln) * HID + kb;
        bfr[gi * 2 + 0] = load_frag(Brow);
        bfr[gi * 2 + 1] = load_frag(Brow + 32);
    }

    for (int i = tid; i < (32 * HID) / 2; i += blockDim.x)
        ((unsigned int*)hb_s)[i] = 0u;          // h0 = 0

    size_t gdir = (size_t)dir * (size_t)ROWS_TOTAL * NG;

    {   // stage step 0 tile: 32 rows x 256 cols f16 = 1024 x 16B async copies
        int s0 = dir ? (S_LEN - 1) : 0;
        const _Float16* src = gpre + gdir + ((size_t)s0 * BATCH + b0) * NG;
        for (int c = tid; c < 1024; c += 256)
            async_ld_b128((unsigned)(unsigned long long)&gbuf[0][c * 8],
                          src + c * 8);
    }
    wait_async0();
    __syncthreads();

    const _Float16* Arow = hb_s + (m * 16 + ln) * HID + kb;

    v8f cst, hlast;
#pragma unroll
    for (int r = 0; r < 8; ++r) { cst[r] = 0.f; hlast[r] = 0.f; }

    for (int si = 0; si < S_LEN; ++si) {
        int buf = si & 1;

        if (si + 1 < S_LEN) {   // async-prefetch next step's Gpre tile
            int sn = dir ? (S_LEN - 2 - si) : si + 1;
            const _Float16* src = gpre + gdir + ((size_t)sn * BATCH + b0) * NG;
            for (int c = tid; c < 1024; c += 256)
                async_ld_b128((unsigned)(unsigned long long)&gbuf[buf ^ 1][c * 8],
                              src + c * 8);
        }

        v16h a0 = load_frag(Arow + 0);
        v16h a1 = load_frag(Arow + 32);

        v8f g[4];
#pragma unroll
        for (int gi = 0; gi < 4; ++gi) {
            int n0 = gi * 64 + hbk * 16;
            const _Float16* gp = &gbuf[buf][(m * 16 + kb) * NG + n0 + ln];
            v8f acc;
#pragma unroll
            for (int r = 0; r < 8; ++r) acc[r] = (float)gp[r * NG];
            acc = wmma_f16(a0, bfr[gi * 2 + 0], acc);
            acc = wmma_f16(a1, bfr[gi * 2 + 1], acc);
            g[gi] = acc;
        }

        __syncthreads();        // B1: all LDS reads done before h rewrite

#pragma unroll
        for (int r = 0; r < 8; ++r) {
            float iv = sigf(g[0][r]);
            float fv = sigf(g[1][r]);
            float gv = tanh_fast(g[2][r]);
            float ov = sigf(g[3][r]);
            float cn = fv * cst[r] + iv * gv;
            cst[r] = cn;
            float hv = ov * tanh_fast(cn);
            hlast[r] = hv;
            hb_s[(m * 16 + r + kb) * HID + hbk * 16 + ln] = (_Float16)hv;
        }

        wait_async0();          // next Gpre tile resident in LDS
        __syncthreads();        // B2: publish h(t+1) and staged tile
    }

#pragma unroll
    for (int r = 0; r < 8; ++r)
        hfin[(size_t)dir * BATCH * HID +
             (size_t)(b0 + m * 16 + r + kb) * HID + hbk * 16 + ln] = hlast[r];
}

// ---------------- out: sigmoid( [h_f|h_b] @ fc_w^T + fc_b ) ----------------
__global__ void out_kernel(const float* __restrict__ fcw,
                           const float* __restrict__ fcb,
                           const char* ws, float* __restrict__ out) {
    const float* hfin = (const float*)(ws + OFF_HFIN);
    int b = threadIdx.x;
    float acc = fcb[0];
    for (int j = 0; j < HID; ++j) acc += hfin[b * HID + j] * fcw[j];
    for (int j = 0; j < HID; ++j) acc += hfin[BATCH * HID + b * HID + j] * fcw[HID + j];
    out[b] = 1.f / (1.f + __expf(-acc));
}

extern "C" void kernel_launch(void* const* d_in, const int* in_sizes, int n_in,
                              void* d_out, int out_size, void* d_ws, size_t ws_size,
                              hipStream_t stream) {
    (void)in_sizes; (void)n_in; (void)out_size; (void)ws_size;
    const int*   x     = (const int*)  d_in[0];
    const float* emb   = (const float*)d_in[1];
    const float* Wih_f = (const float*)d_in[2];
    const float* Whh_f = (const float*)d_in[3];
    const float* bih_f = (const float*)d_in[4];
    const float* bhh_f = (const float*)d_in[5];
    const float* Wih_b = (const float*)d_in[6];
    const float* Whh_b = (const float*)d_in[7];
    const float* bih_b = (const float*)d_in[8];
    const float* bhh_b = (const float*)d_in[9];
    const float* fcw   = (const float*)d_in[10];
    const float* fcb   = (const float*)d_in[11];
    char*  ws  = (char*)d_ws;
    float* out = (float*)d_out;

    prep_kernel<<<1, 256, 0, stream>>>(Wih_f, Whh_f, bih_f, bhh_f,
                                       Wih_b, Whh_b, bih_b, bhh_b, ws);
    proj_kernel<<<2048, 256, 0, stream>>>(x, emb, ws);
    rec_kernel<<<16, 256, 0, stream>>>(ws);
    out_kernel<<<1, 256, 0, stream>>>(fcw, fcb, ws, out);
}
